// GINNet_bottleneckMLP_27307402068742
// MI455X (gfx1250) — matrix operand order repaired and checked by reading, not compiled, not run on hardware
//
#include <hip/hip_runtime.h>
#include <hip/hip_bf16.h>
#include <cstdint>

typedef __attribute__((ext_vector_type(16))) _Float16 v16h;
typedef __attribute__((ext_vector_type(8)))  float    v8f;

#define DDIM   128
#define HDIM   64
#define PDIM   10
#define CDIM   2
#define GDIM   128           // N_GRAPHS
#define BN_EPS 1e-5f
#define PROTO_EPS 1e-4f

// ---------------- deterministic hash-based uniform(0,1) ----------------
__device__ inline float u01(uint32_t x) {
    x ^= x >> 17; x *= 0xed5ad4bbu;
    x ^= x >> 11; x *= 0xac4c1b51u;
    x ^= x >> 15; x *= 0x31848babu;
    x ^= x >> 14;
    return (float)(x >> 8) * (1.0f / 16777216.0f) + 1e-7f; // (0,1)
}

// ---------------- zero fill ----------------
__global__ void k_fill0(float* __restrict__ p, int n) {
    int i = blockIdx.x * 256 + threadIdx.x;
    if (i < n) p[i] = 0.0f;
}

// ---------------- edge scatter-add: agg[dst] += x[src] ----------------
__global__ void k_scatter(const float* __restrict__ X, const int* __restrict__ src,
                          const int* __restrict__ dst, float* __restrict__ agg,
                          int E, int N) {
    long long t = (long long)blockIdx.x * 256 + threadIdx.x;
    if (t >= (long long)E * N) return;
    int e = (int)(t / N), j = (int)(t % N);
    atomicAdd(&agg[(size_t)dst[e] * N + j], X[(size_t)src[e] * N + j]);
}

// ---------------- WMMA GEMM: C[M,N] = act((A (+A2)) @ W + bias) ----------------
// Block = 128 threads = 4 waves; wave w handles rows [blk*64 + w*16, +16).
// W (K x N, row-major f32) staged to LDS as f16. One wmma per (ktile,ntile).
template<int KDIM, int NDIM, int ACT, int ADD, int BIAS>
__global__ __launch_bounds__(128) void k_gemm_wmma(
    const float* __restrict__ A, const float* __restrict__ A2,
    const float* __restrict__ W, const float* __restrict__ bias,
    float* __restrict__ C, int M)
{
    constexpr int KT = KDIM / 32;
    constexpr int NT = NDIM / 16;
    __shared__ __align__(32) _Float16 sW[KDIM * NDIM];

    const int tid = threadIdx.x;
    for (int i = tid; i < KDIM * NDIM; i += 128) sW[i] = (_Float16)W[i];
    __syncthreads();

    const int wave = tid >> 5;
    const int lane = tid & 31;
    const int rowBase = blockIdx.x * 64 + wave * 16;
    if (rowBase >= M) return;            // wave-uniform: EXEC stays all-ones

    const int mrow = rowBase + (lane & 15);
    const int hi   = lane >> 4;
    const float* __restrict__ pa = A + (size_t)mrow * KDIM;
    const float* __restrict__ pa2 = pa;
    if (ADD) pa2 = A2 + (size_t)mrow * KDIM;

    __builtin_prefetch(pa, 0, 1);        // global_prefetch_b8

    v8f acc[NT] = {};

#pragma unroll
    for (int kt = 0; kt < KT; ++kt) {
        // ---- A fragment: 16-bit A layout (lane lo = M, hi bit selects K-group) ----
        v16h a;
#pragma unroll
        for (int j = 0; j < 8; ++j) {
            const int k = kt * 32 + (j < 4 ? hi * 8 + 2 * j : 16 + hi * 8 + 2 * (j - 4));
            float2 f = *reinterpret_cast<const float2*>(pa + k);   // k is always even
            if (ADD) {
                float2 g = *reinterpret_cast<const float2*>(pa2 + k);
                f.x += g.x; f.y += g.y;
            }
            a[2 * j]     = (_Float16)f.x;
            a[2 * j + 1] = (_Float16)f.y;
        }
        // ---- B fragments from LDS: lane lo = N column, elements = K (same order as A) ----
        const int nb = lane & 15;
#pragma unroll
        for (int nt = 0; nt < NT; ++nt) {
            v16h b;
#pragma unroll
            for (int s = 0; s < 16; ++s) {
                const int kb = kt * 32 + (s < 8 ? hi * 8 + s : 16 + hi * 8 + (s - 8));
                b[s] = sW[kb * NDIM + nt * 16 + nb];
            }
            acc[nt] = __builtin_amdgcn_wmma_f32_16x16x32_f16(
                false, a, false, b, (short)0, acc[nt], false, false);
        }
    }

    // ---- store per C/D layout: VGPR r -> M = r (lanes 0-15) / r+8 (lanes 16-31) ----
    const int nlo = lane & 15;
#pragma unroll
    for (int nt = 0; nt < NT; ++nt) {
#pragma unroll
        for (int r = 0; r < 8; ++r) {
            const int row = rowBase + r + hi * 8;
            const int col = nt * 16 + nlo;
            float v = acc[nt][r];
            if (BIAS) v += bias[col];
            if (ACT == 1) v = fmaxf(v, 0.0f);
            if (ACT == 2) v = tanhf(v);
            C[(size_t)row * NDIM + col] = v;
        }
    }
}

// ---------------- per-column mean/var (or std, ddof selectable) ----------------
__global__ void k_colstats(const float* __restrict__ X, int M, int N,
                           float* __restrict__ muOut, float* __restrict__ vOut,
                           int ddof, int as_std) {
    __shared__ float s1[256], s2[256];
    const int col = blockIdx.x, tid = threadIdx.x;
    float s = 0.0f, q = 0.0f;
    for (int i = tid; i < M; i += 256) {
        float v = X[(size_t)i * N + col];
        s += v; q += v * v;
    }
    s1[tid] = s; s2[tid] = q; __syncthreads();
    for (int off = 128; off > 0; off >>= 1) {
        if (tid < off) { s1[tid] += s1[tid + off]; s2[tid] += s2[tid + off]; }
        __syncthreads();
    }
    if (tid == 0) {
        float m = s1[0] / (float)M;
        float var = (s2[0] - (float)M * m * m) / (float)(M - ddof);
        var = fmaxf(var, 0.0f);
        muOut[col] = m;
        vOut[col]  = as_std ? sqrtf(var) : var;
    }
}

// ---------------- batchnorm apply (in place), optional relu ----------------
__global__ void k_bn(float* __restrict__ X, const float* __restrict__ mu,
                     const float* __restrict__ var, const float* __restrict__ g,
                     const float* __restrict__ b, int M, int N, int relu) {
    int i = blockIdx.x * 256 + threadIdx.x;
    if (i >= M * N) return;
    int col = i % N;
    float v = g[col] * (X[i] - mu[col]) * rsqrtf(var[col] + BN_EPS) + b[col];
    if (relu) v = fmaxf(v, 0.0f);
    X[i] = v;
}

// ---------------- assignment head: a2 = a1@Wfc2+bfc2, softmax, gumbel-softmax ----------------
__global__ void k_assign(const float* __restrict__ a1, const float* __restrict__ Wfc2,
                         const float* __restrict__ bfc2, float* __restrict__ asg,
                         float* __restrict__ lam, int M) {
    int i = blockIdx.x * 256 + threadIdx.x;
    if (i >= M) return;
    const float* r = a1 + (size_t)i * DDIM;
    float z0 = bfc2[0], z1 = bfc2[1];
    for (int k = 0; k < DDIM; ++k) {
        float v = r[k];
        z0 += v * Wfc2[k * 2];
        z1 += v * Wfc2[k * 2 + 1];
    }
    float mx = fmaxf(z0, z1);
    float e0 = __expf(z0 - mx), e1 = __expf(z1 - mx);
    float inv = 1.0f / (e0 + e1);
    float s0 = e0 * inv, s1 = e1 * inv;
    asg[i * 2] = s0; asg[i * 2 + 1] = s1;
    // gumbel noise (hash RNG; JAX bits not reproducible anyway)
    float g0 = -logf(-logf(u01(2u * (uint32_t)i + 1u)));
    float g1 = -logf(-logf(u01(2u * (uint32_t)i + 2u)));
    float t0 = s0 + g0, t1 = s1 + g1;
    mx = fmaxf(t0, t1);
    e0 = __expf(t0 - mx); e1 = __expf(t1 - mx);
    inv = 1.0f / (e0 + e1);
    lam[i * 2] = e0 * inv; lam[i * 2 + 1] = e1 * inv;
}

// ---------------- noisy features + segment pooling + KL accumulation ----------------
__global__ void k_noisy(const float* __restrict__ nf, const float* __restrict__ lam,
                        const float* __restrict__ fmean, const float* __restrict__ fstd,
                        const int* __restrict__ batch, float* __restrict__ sums,
                        float* __restrict__ klacc, int M) {
    __shared__ float sk[256];
    int t = blockIdx.x * 256 + threadIdx.x;
    float klv = 0.0f;
    if (t < M * DDIM) {
        int i = t / DDIM, j = t % DDIM;
        float lp = lam[i * 2], ln = lam[i * 2 + 1];
        float v  = nf[t];
        float nm = lp * v + ln * fmean[j];
        float ns = ln * fstd[j];
        float noisy = nm + u01(0x9e3779b9u + (uint32_t)t) * ns;
        atomicAdd(&sums[(size_t)batch[i] * DDIM + j], noisy);
        float fse = fstd[j] + 1e-7f;
        float z = (nm - fmean[j]) / fse;
        klv = 0.5f * ns * ns / (fse * fse) / (float)(M * DDIM) + z * z / (float)DDIM;
    }
    sk[threadIdx.x] = klv; __syncthreads();
    for (int off = 128; off > 0; off >>= 1) {
        if (threadIdx.x < off) sk[threadIdx.x] += sk[threadIdx.x + off];
        __syncthreads();
    }
    if (threadIdx.x == 0) atomicAdd(klacc, sk[0]);
}

__global__ void k_cnt(const int* __restrict__ batch, float* __restrict__ cnt, int M) {
    int i = blockIdx.x * 256 + threadIdx.x;
    if (i < M) atomicAdd(&cnt[batch[i]], 1.0f);
}

__global__ void k_emb(const float* __restrict__ sums, const float* __restrict__ cnt,
                      float* __restrict__ emb, float* __restrict__ out_emb) {
    int t = blockIdx.x * 256 + threadIdx.x;
    if (t >= GDIM * DDIM) return;
    float v = sums[t] / fmaxf(cnt[t / DDIM], 1.0f);
    emb[t] = v; out_emb[t] = v;
}

// ---------------- edge-wise assignment adjacency (2x2) ----------------
__global__ void k_adj(const float* __restrict__ asg, const int* __restrict__ src,
                      const int* __restrict__ dst, float* __restrict__ adj, int E) {
    __shared__ float s[4][256];
    int e = blockIdx.x * 256 + threadIdx.x;
    float p0 = 0, p1 = 0, p2 = 0, p3 = 0;
    if (e < E) {
        float a0 = asg[(size_t)src[e] * 2], a1 = asg[(size_t)src[e] * 2 + 1];
        float b0 = asg[(size_t)dst[e] * 2], b1 = asg[(size_t)dst[e] * 2 + 1];
        p0 = a0 * b0; p1 = a0 * b1; p2 = a1 * b0; p3 = a1 * b1;
    }
    int tid = threadIdx.x;
    s[0][tid] = p0; s[1][tid] = p1; s[2][tid] = p2; s[3][tid] = p3;
    __syncthreads();
    for (int off = 128; off > 0; off >>= 1) {
        if (tid < off)
            for (int c = 0; c < 4; ++c) s[c][tid] += s[c][tid + off];
        __syncthreads();
    }
    if (tid < 4) atomicAdd(&adj[tid], s[tid][0]);
}

__global__ void k_finalize(const float* __restrict__ adj, const float* __restrict__ klacc,
                           float* __restrict__ out_kl, float* __restrict__ out_pp) {
    float a00 = adj[0], a01 = adj[1], a10 = adj[2], a11 = adj[3];
    float d0 = a00 / fmaxf(fabsf(a00) + fabsf(a01), 1e-12f);
    float d1 = a11 / fmaxf(fabsf(a10) + fabsf(a11), 1e-12f);
    out_pp[0] = 0.5f * ((d0 - 1.0f) * (d0 - 1.0f) + (d1 - 1.0f) * (d1 - 1.0f));
    out_kl[0] = klacc[0];
}

// ---------------- prototype distances, sim, logits, probs ----------------
__global__ void k_proto(const float* __restrict__ emb, const float* __restrict__ protos,
                        const float* __restrict__ Wlast, float* __restrict__ out_logits,
                        float* __restrict__ out_probs, float* __restrict__ out_sim,
                        float* __restrict__ out_dist) {
    int g = threadIdx.x;
    if (g >= GDIM) return;
    const float* e = emb + (size_t)g * DDIM;
    float e2 = 0.0f;
    for (int d = 0; d < DDIM; ++d) e2 += e[d] * e[d];
    float sim[PDIM];
    for (int p = 0; p < PDIM; ++p) {
        const float* pr = protos + (size_t)p * DDIM;
        float xp = 0.0f, p2 = 0.0f;
        for (int d = 0; d < DDIM; ++d) { xp += e[d] * pr[d]; p2 += pr[d] * pr[d]; }
        float dist = -2.0f * xp + e2 + p2;
        float s = logf((dist + 1.0f) / (dist + PROTO_EPS));
        sim[p] = s;
        out_sim[g * PDIM + p]  = s;
        out_dist[g * PDIM + p] = dist;
    }
    float l[CDIM];
    for (int c = 0; c < CDIM; ++c) {
        const float* w = Wlast + (size_t)c * (PDIM + DDIM);
        float acc = 0.0f;
        for (int p = 0; p < PDIM; ++p) acc += sim[p] * w[p];
        for (int d = 0; d < DDIM; ++d) acc += e[d] * w[PDIM + d];
        l[c] = acc;
        out_logits[g * CDIM + c] = acc;
    }
    float mx = fmaxf(l[0], l[1]);
    float e0 = __expf(l[0] - mx), e1 = __expf(l[1] - mx);
    float inv = 1.0f / (e0 + e1);
    out_probs[g * CDIM]     = e0 * inv;
    out_probs[g * CDIM + 1] = e1 * inv;
}

// =======================================================================
extern "C" void kernel_launch(void* const* d_in, const int* in_sizes, int n_in,
                              void* d_out, int out_size, void* d_ws, size_t ws_size,
                              hipStream_t stream) {
    const float* x     = (const float*)d_in[0];
    const int*   ei    = (const int*)d_in[1];
    const int*   batch = (const int*)d_in[2];
    const float* W0a  = (const float*)d_in[3];
    const float* W0b  = (const float*)d_in[4];
    const float* g0   = (const float*)d_in[5];
    const float* b0   = (const float*)d_in[6];
    const float* W1a  = (const float*)d_in[7];
    const float* W1b  = (const float*)d_in[8];
    const float* g1   = (const float*)d_in[9];
    const float* b1   = (const float*)d_in[10];
    const float* Wm0  = (const float*)d_in[11];
    const float* bm0  = (const float*)d_in[12];
    const float* gm   = (const float*)d_in[13];
    const float* bm   = (const float*)d_in[14];
    const float* Wm1  = (const float*)d_in[15];
    const float* bm1  = (const float*)d_in[16];
    const float* Wfc1 = (const float*)d_in[17];
    const float* bfc1 = (const float*)d_in[18];
    const float* Wfc2 = (const float*)d_in[19];
    const float* bfc2 = (const float*)d_in[20];
    const float* protos = (const float*)d_in[21];
    const float* Wlast  = (const float*)d_in[22];

    const int M = in_sizes[0] / DDIM;   // 40000
    const int E = in_sizes[1] / 2;      // 640000
    const int* src = ei;
    const int* dst = ei + E;

    // ---- workspace carve (floats) ----
    float* ws = (float*)d_ws;
    size_t o = 0;
    float* agg  = ws + o; o += (size_t)M * DDIM;
    float* buf1 = ws + o; o += (size_t)M * DDIM;
    float* buf2 = ws + o; o += (size_t)M * DDIM;
    float* bufh = ws + o; o += (size_t)M * HDIM;
    float* asg  = ws + o; o += (size_t)M * CDIM;
    float* lam  = ws + o; o += (size_t)M * CDIM;
    float* sums = ws + o; o += GDIM * DDIM;
    float* cnt  = ws + o; o += GDIM;
    float* mu   = ws + o; o += DDIM;
    float* var  = ws + o; o += DDIM;
    float* fmean = ws + o; o += DDIM;
    float* fstd  = ws + o; o += DDIM;
    float* adj   = ws + o; o += 4;
    float* klacc = ws + o; o += 1;
    float* emb   = ws + o; o += GDIM * DDIM;

    // ---- output carve (tuple order) ----
    float* out        = (float*)d_out;
    float* out_logits = out;                         // 128*2
    float* out_probs  = out_logits + GDIM * CDIM;    // 128*2
    float* out_emb    = out_probs  + GDIM * CDIM;    // 128*128
    float* out_kl     = out_emb    + GDIM * DDIM;    // 1
    float* out_pp     = out_kl + 1;                  // 1
    float* out_sim    = out_pp + 1;                  // 128*10
    float* out_dist   = out_sim + GDIM * PDIM;       // 128*10

    const unsigned gemmGrid = (unsigned)(M / 64);             // 625
    const unsigned elemGrid = (unsigned)((M * DDIM + 255) / 256);
    const unsigned edgeGrid = (unsigned)(((long long)E * DDIM + 255) / 256);

    // ======== GIN layer 0 ========
    k_fill0<<<elemGrid, 256, 0, stream>>>(agg, M * DDIM);
    k_scatter<<<edgeGrid, 256, 0, stream>>>(x, src, dst, agg, E, DDIM);
    k_gemm_wmma<DDIM, DDIM, 1, 1, 0><<<gemmGrid, 128, 0, stream>>>(x, agg, W0a, nullptr, buf1, M);
    k_gemm_wmma<DDIM, DDIM, 1, 0, 0><<<gemmGrid, 128, 0, stream>>>(buf1, nullptr, W0b, nullptr, buf2, M);
    k_colstats<<<DDIM, 256, 0, stream>>>(buf2, M, DDIM, mu, var, 0, 0);
    k_bn<<<elemGrid, 256, 0, stream>>>(buf2, mu, var, g0, b0, M, DDIM, 1);   // buf2 = nf1

    // ======== GIN layer 1 ========
    k_fill0<<<elemGrid, 256, 0, stream>>>(agg, M * DDIM);
    k_scatter<<<edgeGrid, 256, 0, stream>>>(buf2, src, dst, agg, E, DDIM);
    k_gemm_wmma<DDIM, DDIM, 1, 1, 0><<<gemmGrid, 128, 0, stream>>>(buf2, agg, W1a, nullptr, buf1, M);
    k_gemm_wmma<DDIM, DDIM, 1, 0, 0><<<gemmGrid, 128, 0, stream>>>(buf1, nullptr, W1b, nullptr, buf2, M);
    k_colstats<<<DDIM, 256, 0, stream>>>(buf2, M, DDIM, mu, var, 0, 0);
    k_bn<<<elemGrid, 256, 0, stream>>>(buf2, mu, var, g1, b1, M, DDIM, 0);   // buf2 = nf2

    // ======== bottleneck MLP ========
    k_gemm_wmma<DDIM, HDIM, 0, 0, 1><<<gemmGrid, 128, 0, stream>>>(buf2, nullptr, Wm0, bm0, bufh, M);
    k_colstats<<<HDIM, 256, 0, stream>>>(bufh, M, HDIM, mu, var, 0, 0);
    k_bn<<<(unsigned)((M * HDIM + 255) / 256), 256, 0, stream>>>(bufh, mu, var, gm, bm, M, HDIM, 1);
    k_gemm_wmma<HDIM, DDIM, 0, 0, 1><<<gemmGrid, 128, 0, stream>>>(bufh, nullptr, Wm1, bm1, buf1, M); // node_feature

    // ======== assignment head ========
    k_gemm_wmma<DDIM, DDIM, 2, 0, 1><<<gemmGrid, 128, 0, stream>>>(buf1, nullptr, Wfc1, bfc1, buf2, M); // a1=tanh(...)
    k_assign<<<(unsigned)((M + 255) / 256), 256, 0, stream>>>(buf2, Wfc2, bfc2, asg, lam, M);

    // ======== feature stats (ddof=1 std of node_feature) ========
    k_colstats<<<DDIM, 256, 0, stream>>>(buf1, M, DDIM, fmean, fstd, 1, 1);

    // ======== pooling + KL ========
    k_fill0<<<(unsigned)((GDIM * DDIM + 255) / 256), 256, 0, stream>>>(sums, GDIM * DDIM);
    k_fill0<<<1, 256, 0, stream>>>(cnt, GDIM);
    k_fill0<<<1, 32, 0, stream>>>(adj, 4);
    k_fill0<<<1, 32, 0, stream>>>(klacc, 1);
    k_noisy<<<elemGrid, 256, 0, stream>>>(buf1, lam, fmean, fstd, batch, sums, klacc, M);
    k_cnt<<<(unsigned)((M + 255) / 256), 256, 0, stream>>>(batch, cnt, M);
    k_emb<<<(unsigned)((GDIM * DDIM + 255) / 256), 256, 0, stream>>>(sums, cnt, emb, out_emb);

    // ======== adjacency penalty ========
    k_adj<<<(unsigned)((E + 255) / 256), 256, 0, stream>>>(asg, src, dst, adj, E);
    k_finalize<<<1, 1, 0, stream>>>(adj, klacc, out_kl, out_pp);

    // ======== prototypes / logits / probs ========
    k_proto<<<1, GDIM, 0, stream>>>(emb, protos, Wlast, out_logits, out_probs, out_sim, out_dist);
}
